// CoLayer_50568944943603
// MI455X (gfx1250) — matrix operand order repaired and checked by reading, not compiled, not run on hardware
//
#include <hip/hip_runtime.h>

typedef __attribute__((ext_vector_type(16))) __bf16 v16bf;
typedef __attribute__((ext_vector_type(8)))  float  v8f;
typedef __attribute__((ext_vector_type(4)))  unsigned int v4u;
typedef __attribute__((ext_vector_type(8)))  int v8i;
typedef __attribute__((ext_vector_type(4)))  int v4i;

#define BATCH 8
#define SEQ   2048
#define DIM   128
#define ROWS_PER_BLOCK 128
#define JT    64
#define NT    (SEQ / JT)
#define LOG2E 1.4426950408889634f

union BFrag {
    v16bf v;
    unsigned short s[16];
    uint4 q[2];
};

__device__ inline unsigned short f2bf(float f) {
    return __builtin_bit_cast(unsigned short, (__bf16)f);   // RNE fptrunc
}

// ---------------- Kernel 1: y (f32) -> yb (bf16, row major) + ybT (bf16, transposed)
__global__ __launch_bounds__(256) void convert_y_kernel(
    const float* __restrict__ y, unsigned short* __restrict__ yb,
    unsigned short* __restrict__ ybT) {
    const int t    = blockIdx.x * 256 + threadIdx.x;  // 0 .. B*S*(D/8)-1
    const int dseg = t & 15;                          // D/8 = 16 segments
    const int s    = (t >> 4) & (SEQ - 1);
    const int b    = t >> 15;                         // 16*2048 per batch
    const size_t rowoff = (size_t)(b * SEQ + s) * DIM + dseg * 8;
    const float4* src = (const float4*)(y + rowoff);
    const float4 f0 = src[0], f1 = src[1];
    unsigned short h[8];
    h[0] = f2bf(f0.x); h[1] = f2bf(f0.y); h[2] = f2bf(f0.z); h[3] = f2bf(f0.w);
    h[4] = f2bf(f1.x); h[5] = f2bf(f1.y); h[6] = f2bf(f1.z); h[7] = f2bf(f1.w);
    union { unsigned short u16[8]; uint4 q; } pk;
#pragma unroll
    for (int i = 0; i < 8; ++i) pk.u16[i] = h[i];
    *(uint4*)(yb + rowoff) = pk.q;                    // coalesced row-major copy
    const size_t tbase = (size_t)(b * DIM + dseg * 8) * SEQ + s;
#pragma unroll
    for (int i = 0; i < 8; ++i) ybT[tbase + (size_t)i * SEQ] = h[i];
}

// ---------------- TDM: 2-D tile DMA global(bf16) -> LDS with row padding ----
__device__ inline void tdm_load_2d(unsigned lds_addr, const unsigned short* gaddr,
                                   int tensor_d0, int tensor_d1,
                                   int tile_d0, int tile_d1,
                                   long long stride0,
                                   int pad_interval_code, int pad_amount_code) {
    const unsigned long long ga = (unsigned long long)(uintptr_t)gaddr;
    v4u g0;
    g0[0] = 1u;                                            // count=1, user mode
    g0[1] = lds_addr;                                      // LDS byte address
    g0[2] = (unsigned)(ga & 0xFFFFFFFFull);                // global_addr[31:0]
    g0[3] = ((unsigned)(ga >> 32) & 0x01FFFFFFu) | (2u << 30);  // [56:32] | type=2
    v8i g1;
    g1[0] = (1 << 16)                                      // data_size = 2 bytes
          | (1 << 20)                                      // pad_enable
          | (pad_interval_code << 22) | (pad_amount_code << 25);
    g1[1] = (tensor_d0 & 0xFFFF) << 16;                    // tensor_dim0[15:0]
    g1[2] = ((tensor_d0 >> 16) & 0xFFFF) | ((tensor_d1 & 0xFFFF) << 16);
    g1[3] = ((tensor_d1 >> 16) & 0xFFFF) | (tile_d0 << 16);
    g1[4] = tile_d1 & 0xFFFF;                              // tile_dim1 (tile_dim2=0)
    g1[5] = (int)(stride0 & 0xFFFFFFFFll);                 // tensor_dim0_stride lo
    g1[6] = (int)((stride0 >> 32) & 0xFFFFll);             // stride hi (dim1_stride=0)
    g1[7] = 0;
    const v4i z4 = {0, 0, 0, 0};
    const v8i z8 = {0, 0, 0, 0, 0, 0, 0, 0};
    __builtin_amdgcn_tensor_load_to_lds(g0, g1, z4, z4, z8, 0);
}

// ---------------- Kernel 2: flash attention + residual ----------------
__global__ __launch_bounds__(256) void flash_attn_residual_kernel(
    const float* __restrict__ x, const unsigned short* __restrict__ yb,
    const unsigned short* __restrict__ ybT, float* __restrict__ out) {
    // double-buffered y tile, bf16, row-major [key][d]  (A-operand, S^T = Y*X^T)
    __shared__ __align__(16) unsigned short lds_y[2][JT][DIM + 8];
    // double-buffered y tile, bf16, transposed [d][key] (B-operand, O += P*Y)
    __shared__ __align__(16) unsigned short lds_yT[2][DIM][JT + 8];

    const int tid    = threadIdx.x;
    const int wave   = tid >> 5;
    const int lane   = tid & 31;
    const int lane16 = lane & 15;
    const int hi     = lane >> 4;
    const int b      = blockIdx.y;
    const int i0     = blockIdx.x * ROWS_PER_BLOCK + wave * 16;

    // ---- X fragments in B-layout (lane = x-row, elems = 16 contiguous d),
    //      pre-scaled by log2(e) so scores are in log2 units ----
    const float* xrow = x + (size_t)(b * SEQ + i0 + lane16) * DIM;
    BFrag xb[4];
#pragma unroll
    for (int kf = 0; kf < 4; ++kf) {
        const int kb = 32 * kf + 16 * hi;
#pragma unroll
        for (int i = 0; i < 16; ++i)
            xb[kf].s[i] = f2bf(xrow[kb + i] * LOG2E);
    }

    const v8f vzero = {0.f, 0.f, 0.f, 0.f, 0.f, 0.f, 0.f, 0.f};
    float m = -__builtin_inff();
    float l = 0.0f;
    v8f o[8];
#pragma unroll
    for (int t = 0; t < 8; ++t) o[t] = vzero;

    const unsigned short* ybb  = yb  + (size_t)b * SEQ * DIM;
    const unsigned short* ybTb = ybT + (size_t)b * DIM * SEQ;

    auto issue_tile = [&](int nbuf, int j0) {
        // y tile: rows = 64 keys of 128 d, stride DIM; pad 4DW per 64DW row
        tdm_load_2d((unsigned)(uintptr_t)&lds_y[nbuf][0][0],
                    ybb + (size_t)j0 * DIM, DIM, JT, DIM, JT, DIM, 5, 3);
        // yT tile: rows = 128 d of 64 keys, stride SEQ; pad 4DW per 32DW row
        tdm_load_2d((unsigned)(uintptr_t)&lds_yT[nbuf][0][0],
                    ybTb + j0, JT, DIM, JT, DIM, SEQ, 4, 3);
    };

    auto compute_tile = [&](int buf) {
        // ---- S^T(64x16) = Y(64x128) * X^T(128x16), f32 accum ----
        v8f st[4];
#pragma unroll
        for (int t = 0; t < 4; ++t) st[t] = vzero;
#pragma unroll
        for (int t = 0; t < 4; ++t) {
#pragma unroll
            for (int kf = 0; kf < 4; ++kf) {
                BFrag a;   // A-layout: lane = key row 16t+lane16
                const unsigned short* pa =
                    &lds_y[buf][16 * t + lane16][32 * kf + 16 * hi];
                a.q[0] = *(const uint4*)(pa);
                a.q[1] = *(const uint4*)(pa + 8);
                st[t] = __builtin_amdgcn_wmma_f32_16x16x32_bf16(
                    false, a.v, false, xb[kf].v, (short)0, st[t], false, false);
            }
        }

        // ---- online softmax: this lane owns x-row = lane16 ----
        v8f pm;
#pragma unroll
        for (int r = 0; r < 8; ++r)
            pm[r] = fmaxf(fmaxf(st[0][r], st[1][r]), fmaxf(st[2][r], st[3][r]));
        float mx = fmaxf(fmaxf(fmaxf(pm[0], pm[1]), fmaxf(pm[2], pm[3])),
                         fmaxf(fmaxf(pm[4], pm[5]), fmaxf(pm[6], pm[7])));
        mx = fmaxf(mx, __shfl_xor(mx, 16, 32));
        const float mnew  = fmaxf(m, mx);
        const float scale = __builtin_amdgcn_exp2f(m - mnew);
        m = mnew;
#pragma unroll
        for (int t = 0; t < 4; ++t)
#pragma unroll
            for (int r = 0; r < 8; ++r)
                st[t][r] = __builtin_amdgcn_exp2f(st[t][r] - mnew);
        v8f ps;
#pragma unroll
        for (int r = 0; r < 8; ++r)
            ps[r] = (st[0][r] + st[1][r]) + (st[2][r] + st[3][r]);
        float rs = ((ps[0] + ps[1]) + (ps[2] + ps[3])) +
                   ((ps[4] + ps[5]) + (ps[6] + ps[7]));
        rs += __shfl_xor(rs, 16, 32);
        l = l * scale + rs;

        // ---- rescale O: broadcast per-row scale (lane=row) to C-elements ----
        float sc[8];
#pragma unroll
        for (int r = 0; r < 8; ++r) sc[r] = __shfl(scale, r + 8 * hi, 32);
#pragma unroll
        for (int t2 = 0; t2 < 8; ++t2)
#pragma unroll
            for (int r = 0; r < 8; ++r) o[t2][r] *= sc[r];

        // ---- P fragments: C-layout of S^T IS the A-layout of P ----
        BFrag pfr[2];
#pragma unroll
        for (int mf = 0; mf < 2; ++mf)
#pragma unroll
            for (int i = 0; i < 8; ++i) {
                pfr[mf].s[i]     = f2bf(st[2 * mf][i]);
                pfr[mf].s[i + 8] = f2bf(st[2 * mf + 1][i]);
            }

        // ---- O(16x128) += P(16x64) * Y(64x128) ----
#pragma unroll
        for (int t2 = 0; t2 < 8; ++t2) {
#pragma unroll
            for (int mf = 0; mf < 2; ++mf) {
                BFrag bb;  // B-layout: lane = d-col 16*t2+lane16
                const unsigned short* pb =
                    &lds_yT[buf][16 * t2 + lane16][32 * mf + 8 * hi];
                bb.q[0] = *(const uint4*)(pb);
                bb.q[1] = *(const uint4*)(pb + 16);
                o[t2] = __builtin_amdgcn_wmma_f32_16x16x32_bf16(
                    false, pfr[mf].v, false, bb.v, (short)0, o[t2], false, false);
            }
        }
    };

    // ---- prologue: DMA tile 0 ----
    if (wave == 0) {
        issue_tile(0, 0);
        __builtin_amdgcn_s_wait_tensorcnt((short)0);
    }
    __syncthreads();

    // ---- main loop: TDM for tile j+1 overlaps WMMA on tile j ----
    for (int jt = 0; jt < NT; ++jt) {
        const int buf = jt & 1;
        if (wave == 0 && jt + 1 < NT) issue_tile(buf ^ 1, (jt + 1) * JT);
        compute_tile(buf);
        if (wave == 0) __builtin_amdgcn_s_wait_tensorcnt((short)0);
        __syncthreads();
    }

    // ---- epilogue: out = x + O / l ----
    const float rl = __builtin_amdgcn_rcpf(l);
    float rls[8];
#pragma unroll
    for (int r = 0; r < 8; ++r) rls[r] = __shfl(rl, r + 8 * hi, 32);
#pragma unroll
    for (int r = 0; r < 8; ++r) {
        const int row = i0 + r + 8 * hi;
        const size_t base = (size_t)(b * SEQ + row) * DIM;
#pragma unroll
        for (int t2 = 0; t2 < 8; ++t2) {
            const size_t idx = base + 16 * t2 + lane16;
            out[idx] = x[idx] + o[t2][r] * rls[r];
        }
    }
}

extern "C" void kernel_launch(void* const* d_in, const int* in_sizes, int n_in,
                              void* d_out, int out_size, void* d_ws, size_t ws_size,
                              hipStream_t stream) {
    (void)in_sizes; (void)n_in; (void)out_size; (void)ws_size;
    const float* x = (const float*)d_in[0];
    const float* y = (const float*)d_in[1];
    float* out = (float*)d_out;
    unsigned short* yb  = (unsigned short*)d_ws;
    unsigned short* ybT = yb + (size_t)BATCH * SEQ * DIM;

    convert_y_kernel<<<(BATCH * SEQ * (DIM / 8)) / 256, 256, 0, stream>>>(y, yb, ybT);

    dim3 grid(SEQ / ROWS_PER_BLOCK, BATCH);
    flash_attn_residual_kernel<<<grid, 256, 0, stream>>>(x, yb, ybT, out);
}
